// PhyloSpec_57707180589354
// MI455X (gfx1250) — compile-verified, matrix operand order checked by reading, not run on hardware
//
#include <hip/hip_runtime.h>

#define B_    256
#define L_    1024
#define C_    16
#define F_    90112      // C * 11*L/2
#define POOL_ 5632       // pooled length per channel
#define KS_   8          // split-K factor for fc1 GEMM
#define KC_   (F_ / KS_) // 11264 K per split

typedef __attribute__((ext_vector_type(16))) __bf16 v16bf;
typedef __attribute__((ext_vector_type(8)))  float  v8f;

__device__ __forceinline__ unsigned short f2bf(float f) {
  union { float f; unsigned int u; } v; v.f = f;
  unsigned int u = v.u;
  return (unsigned short)((u + 0x7FFFu + ((u >> 16) & 1u)) >> 16); // RNE
}

// chunk layout: chunk[b][c][p] at ((b*16+c)*1024 + p)
__global__ void leaf_kernel(const float* __restrict__ x, const float* __restrict__ lw,
                            const float* __restrict__ lb, const float* __restrict__ nw,
                            float* __restrict__ chunk0) {
  int g = blockIdx.x * 256 + threadIdx.x;          // B*C*L threads
  int b = g >> 14; int rem = g & 16383; int c = rem >> 10; int i = rem & 1023;
  float v = (x[(b << 10) + i] * lw[(i << 4) + c] + lb[(i << 4) + c]) * nw[i];
  chunk0[g] = v;
}

// One conv level: per-segment SAME conv (k=3), bias, BN, ReLU, node-weight scale.
__global__ __launch_bounds__(256) void conv_kernel(
    const float* __restrict__ src, float* __restrict__ dst,
    const float* __restrict__ cw, const float* __restrict__ cb,
    const float* __restrict__ gam, const float* __restrict__ bet,
    const float* __restrict__ nw, int level, int nwOff) {
  __shared__ float sw[768];                 // [co][ci][k] = co*48+ci*3+k (matches OIH)
  __shared__ float sb[16], ssc[16], sbe[16];
  __shared__ float sin_[16][260];
  int tid  = threadIdx.x;
  int b    = blockIdx.y;
  int pos0 = blockIdx.x << 8;
  for (int idx = tid; idx < 768; idx += 256) sw[idx] = cw[idx];
  if (tid < 16) {
    sb[tid]  = cb[tid];
    ssc[tid] = gam[tid] * rsqrtf(1.0f + 1e-5f);
    sbe[tid] = bet[tid];
  }
  #pragma unroll
  for (int c = 0; c < 16; ++c)
    sin_[c][1 + tid] = src[((b << 4) + c) * 1024 + pos0 + tid];
  if (tid < 16) {
    sin_[tid][0] = (pos0 > 0) ? src[((b << 4) + tid) * 1024 + pos0 - 1] : 0.0f;
  } else if (tid < 32) {
    int c = tid - 16;
    sin_[c][257] = (pos0 + 256 < 1024) ? src[((b << 4) + c) * 1024 + pos0 + 256] : 0.0f;
  }
  __syncthreads();

  int pos = pos0 + tid;
  int S   = 1 << level;
  int t   = pos & (S - 1);
  int m   = pos >> level;
  float nodeW = nw[nwOff + m];
  bool hasL = (t != 0), hasR = (t != S - 1);

  float acc[16];
  #pragma unroll
  for (int co = 0; co < 16; ++co) acc[co] = 0.0f;
  #pragma unroll
  for (int ci = 0; ci < 16; ++ci) {
    float lv = hasL ? sin_[ci][tid]     : 0.0f;
    float mv =        sin_[ci][tid + 1];
    float rv = hasR ? sin_[ci][tid + 2] : 0.0f;
    #pragma unroll
    for (int co = 0; co < 16; ++co) {
      const float* w = &sw[co * 48 + ci * 3];
      acc[co] += lv * w[0] + mv * w[1] + rv * w[2];
    }
  }
  #pragma unroll
  for (int co = 0; co < 16; ++co) {
    float v = (acc[co] + sb[co]) * ssc[co] + sbe[co];
    v = fmaxf(v, 0.0f) * nodeW;
    dst[((b << 4) + co) * 1024 + pos] = v;
  }
}

// Pairwise max-pool of one 1024-chunk into flat bf16 (B, F): k = c*5632 + j*512 + u
__global__ void pool_kernel(const float* __restrict__ chunk,
                            unsigned short* __restrict__ flat, int j) {
  int g = blockIdx.x * 256 + threadIdx.x;          // B*C*512 threads
  int b = g >> 13; int rem = g & 8191; int c = rem >> 9; int u = rem & 511;
  const float* p = chunk + ((b << 4) + c) * 1024 + (u << 1);
  flat[(size_t)b * F_ + c * POOL_ + j * 512 + u] = f2bf(fmaxf(p[0], p[1]));
}

__global__ void cvt_kernel(const float* __restrict__ w, unsigned short* __restrict__ o, int n) {
  int g = blockIdx.x * 256 + threadIdx.x;
  if (g < n) o[g] = f2bf(w[g]);
}

// fc1 GEMM: out[b,n] = sum_k flat[b,k]*w1[n,k].  One wave = one 16x16 tile, split-K.
// Lanes 0-15 load K[kk..kk+15], lanes 16-31 load K[kk+16..kk+31]; identical packing
// for A and B -> K-permutation-invariant WMMA result.
__global__ __launch_bounds__(32) void gemm_kernel(const unsigned short* __restrict__ Abf,
                                                  const unsigned short* __restrict__ Bbf,
                                                  float* __restrict__ part) {
  int lane = threadIdx.x;
  int row  = lane & 15;
  int hi   = lane >> 4;
  int mT = blockIdx.x, nT = blockIdx.y, z = blockIdx.z;
  size_t aBase = (size_t)(mT * 16 + row) * F_ + (size_t)z * KC_ + 16 * hi;
  size_t bBase = (size_t)(nT * 16 + row) * F_ + (size_t)z * KC_ + 16 * hi;
  v8f acc = {};
  for (int kk = 0; kk < KC_; kk += 32) {
    v16bf a = *(const v16bf*)(Abf + aBase + kk);
    v16bf b = *(const v16bf*)(Bbf + bBase + kk);
    acc = __builtin_amdgcn_wmma_f32_16x16x32_bf16(false, a, false, b,
                                                  (short)0, acc, false, false);
  }
  // C/D layout: VGPR r -> M = r + 8*hi, N = lane&15
  float* out = part + ((size_t)z * 256 + mT * 16 + 8 * hi) * 128 + nT * 16 + row;
  #pragma unroll
  for (int r = 0; r < 8; ++r) out[(size_t)r * 128] = acc[r];
}

__global__ void reduce_kernel(const float* __restrict__ part, const float* __restrict__ b1,
                              float* __restrict__ h1) {
  int g = blockIdx.x * 256 + threadIdx.x;          // 256*128 threads
  float s = 0.0f;
  #pragma unroll
  for (int z = 0; z < KS_; ++z) s += part[(size_t)z * 32768 + g];
  s += b1[g & 127];
  h1[g] = fmaxf(s, 0.0f);
}

__global__ void fc2_kernel(const float* __restrict__ h1, const float* __restrict__ w2,
                           const float* __restrict__ b2, float* __restrict__ h2) {
  int g = blockIdx.x * 256 + threadIdx.x;          // 256*64 threads
  int b = g >> 6; int n = g & 63;
  const float* hp = h1 + b * 128;
  const float* wp = w2 + n * 128;
  float s = 0.0f;
  #pragma unroll 8
  for (int k = 0; k < 128; ++k) s += hp[k] * wp[k];
  h2[g] = fmaxf(s + b2[n], 0.0f);
}

__global__ void fc3_kernel(const float* __restrict__ h2, const float* __restrict__ w3,
                           const float* __restrict__ b3, float* __restrict__ out) {
  int b = threadIdx.x;                              // 256 threads
  const float* hp = h2 + b * 64;
  float s = 0.0f;
  #pragma unroll 8
  for (int k = 0; k < 64; ++k) s += hp[k] * w3[k];
  out[b] = s + b3[0];
}

extern "C" void kernel_launch(void* const* d_in, const int* in_sizes, int n_in,
                              void* d_out, int out_size, void* d_ws, size_t ws_size,
                              hipStream_t stream) {
  (void)in_sizes; (void)n_in; (void)out_size; (void)ws_size;
  const float* x   = (const float*)d_in[0];
  const float* lw  = (const float*)d_in[1];
  const float* lb  = (const float*)d_in[2];
  const float* cw  = (const float*)d_in[3];
  const float* cb  = (const float*)d_in[4];
  const float* gam = (const float*)d_in[5];
  const float* bet = (const float*)d_in[6];
  const float* nw  = (const float*)d_in[7];
  const float* w1  = (const float*)d_in[8];
  const float* b1  = (const float*)d_in[9];
  const float* w2  = (const float*)d_in[10];
  const float* b2  = (const float*)d_in[11];
  const float* w3  = (const float*)d_in[12];
  const float* b3  = (const float*)d_in[13];

  char* ws = (char*)d_ws;
  size_t off = 0;
  auto alloc = [&](size_t bytes) {
    void* p = ws + off;
    off = (off + bytes + 255) & ~(size_t)255;
    return p;
  };
  float*          chunkA = (float*)alloc(sizeof(float) * B_ * C_ * L_);          // 16.8 MB
  float*          chunkB = (float*)alloc(sizeof(float) * B_ * C_ * L_);          // 16.8 MB
  unsigned short* flatbf = (unsigned short*)alloc(2ull * B_ * F_);               // 46 MB
  unsigned short* w1bf   = (unsigned short*)alloc(2ull * 128 * F_);              // 23 MB
  float*          part   = (float*)alloc(sizeof(float) * KS_ * 256 * 128);       // 1 MB
  float*          h1     = (float*)alloc(sizeof(float) * 256 * 128);
  float*          h2     = (float*)alloc(sizeof(float) * 256 * 64);

  leaf_kernel<<<(B_ * C_ * L_) / 256, 256, 0, stream>>>(x, lw, lb, nw, chunkA);
  pool_kernel<<<(B_ * C_ * 512) / 256, 256, 0, stream>>>(chunkA, flatbf, 0);

  float* src = chunkA; float* dst = chunkB;
  int nwOff = L_;
  for (int j = 1; j <= 10; ++j) {
    conv_kernel<<<dim3(4, B_), 256, 0, stream>>>(src, dst, cw, cb, gam, bet, nw, j, nwOff);
    pool_kernel<<<(B_ * C_ * 512) / 256, 256, 0, stream>>>(dst, flatbf, j);
    nwOff += L_ >> j;
    float* tmp = src; src = dst; dst = tmp;
  }

  cvt_kernel<<<(128 * F_) / 256, 256, 0, stream>>>(w1, w1bf, 128 * F_);
  gemm_kernel<<<dim3(16, 8, KS_), 32, 0, stream>>>(flatbf, w1bf, part);
  reduce_kernel<<<128, 256, 0, stream>>>(part, b1, h1);
  fc2_kernel<<<64, 256, 0, stream>>>(h1, w2, b2, h2);
  fc3_kernel<<<1, 256, 0, stream>>>(h2, w3, b3, (float*)d_out);
}
